// MPNNProcessor_82652350644599
// MI455X (gfx1250) — compile-verified, matrix operand order checked by reading, not run on hardware
//
#include <hip/hip_runtime.h>
#include <hip/hip_bf16.h>
#include <math.h>

#define Bb 4
#define Nn 256
#define Hh 128
#define Tt 8
#define BNH (Bb*Nn*Hh)

typedef __attribute__((ext_vector_type(16))) __bf16 v16bf;
typedef __attribute__((ext_vector_type(8)))  __bf16 bf8;
typedef __attribute__((ext_vector_type(8)))  float  v8f;

union frag_u { v16bf v; bf8 h[2]; };

// Native f32->bf16 conversion (gfx1250 has hardware cvt; clang lowers the
// cast to it with round-to-nearest-even). Replaces the manual integer
// rounding sequence that was costing ~5 VALU ops per element.
__device__ __forceinline__ __bf16 f2bf(float f) {
  return (__bf16)f;
}

// Fragment gather per CDNA5 16-bit A-matrix 16x32 layout:
// lanes 0-15 hold K = {kt*32+0..7, kt*32+16..23}, lanes 16-31 the +8 halves.
// B (32x16) is read from a transposed [n][k] LDS tile with the same pattern.
__device__ __forceinline__ v16bf load_frag(const __bf16* tile, int row, int kt, int lane) {
  const int LDK = 136;                         // padded k-stride (halves)
  int koff = kt * 32 + ((lane >> 4) << 3);
  const __bf16* p = tile + row * LDK + koff;
  frag_u f;
  f.h[0] = *(const bf8*)(p);                   // ds_load_b128
  f.h[1] = *(const bf8*)(p + 16);              // ds_load_b128
  return f.v;
}

// ---------------------------------------------------------------------------
// K1: u = z@Wzi + g@Wg + bm1 ; v = z@Wzj + e@We   (per (b,n), 128 threads)
// ---------------------------------------------------------------------------
__global__ __launch_bounds__(128) void uv_kernel(
    const float* __restrict__ node, const float* __restrict__ prev,
    const float* __restrict__ edge, const float* __restrict__ graph,
    const float* __restrict__ Wm1,  const float* __restrict__ bm1,
    float* __restrict__ u, float* __restrict__ v)
{
  __shared__ float sn[128], sp[128], se[128], sg[128];
  int bn = blockIdx.x, t = threadIdx.x;
  int b = bn >> 8;                              // N == 256
  sn[t] = node[bn*Hh + t];
  sp[t] = prev[bn*Hh + t];
  se[t] = edge[bn*Hh + t];
  sg[t] = graph[b*Hh + t];
  __syncthreads();
  float uu = bm1[t], vv = 0.0f;
  for (int k = 0; k < Hh; ++k) {
    uu += sn[k]*Wm1[k*Hh + t] + sp[k]*Wm1[(Hh + k)*Hh + t]
        + sg[k]*Wm1[(5*Hh + k)*Hh + t];
    vv += sn[k]*Wm1[(2*Hh + k)*Hh + t] + sp[k]*Wm1[(3*Hh + k)*Hh + t]
        + se[k]*Wm1[(4*Hh + k)*Hh + t];
  }
  u[bn*Hh + t] = uu;
  v[bn*Hh + t] = vv;
}

// ---------------------------------------------------------------------------
// K2 (dominant): per (b,i): m_i = max_j( relu(relu(u_i+v_j)@Wm2+bm2)@Wm3 ) + bm3
// 256 threads = 8 waves; bf16 WMMA 16x16x32, fp32 accumulate, fully LDS-resident.
// ---------------------------------------------------------------------------
__global__ __launch_bounds__(256) void pair_msgs_kernel(
    const float* __restrict__ u,   const float* __restrict__ v,
    const float* __restrict__ Wm2, const float* __restrict__ bm2,
    const float* __restrict__ Wm3, const float* __restrict__ bm3,
    float* __restrict__ m_out)
{
  extern __shared__ char smem[];
  const int LDK = 136;
  __bf16* Wm2T = (__bf16*)smem;                 // 128 x 136 halves
  __bf16* Wm3T = Wm2T + 128*LDK;
  __bf16* h1s  = Wm3T + 128*LDK;                // 128-row j-strip of relu(u+v)
  __bf16* h2s  = h1s  + 128*LDK;                // GEMM1 output strip
  float*  us    = (float*)(h2s + 128*LDK);      // u_i          (128)
  float*  bm2s  = us + 128;                     // bm2          (128)
  float*  mpart = bm2s + 128;                   // 8 waves x 2 halves x 128
  float*  mbuf  = mpart + 8*2*128;              // running column max (128)

  const int tid   = threadIdx.x;
  const int lane  = tid & 31;
  const int wv    = tid >> 5;
  const int bn    = blockIdx.x;                 // b*N + i
  const int b     = bn >> 8;
  const int base_m = wv * 16;
  const float* vB = v + b*Nn*Hh;

  // Stage weights transposed [n][k] as bf16 (coalesced global reads, strided LDS writes)
  for (int idx = tid; idx < 128*128; idx += 256) {
    int k = idx >> 7, n = idx & 127;
    Wm2T[n*LDK + k] = f2bf(Wm2[idx]);
    Wm3T[n*LDK + k] = f2bf(Wm3[idx]);
  }
  if (tid < 128) {
    us[tid]   = u[bn*Hh + tid];
    bm2s[tid] = bm2[tid];
    mbuf[tid] = -3.0e38f;
  }
  __syncthreads();

  for (int jt = 0; jt < 2; ++jt) {              // two strips of 128 j-rows
    { // build h1 strip: thread handles half a row
      int jl = tid >> 1;
      int c0 = (tid & 1) * 64;
      const float* vr = vB + (jt*128 + jl) * Hh;
      __bf16* hr = h1s + jl * LDK;
      #pragma unroll 8
      for (int c = 0; c < 64; ++c) {
        float x = us[c0 + c] + vr[c0 + c];
        hr[c0 + c] = f2bf(fmaxf(x, 0.0f));
      }
    }
    __syncthreads();

    { // GEMM1: h2 = relu(h1 @ Wm2 + bm2)  (each wave: 16 rows x 128 cols)
      v16bf afr[4];
      #pragma unroll
      for (int kt = 0; kt < 4; ++kt)
        afr[kt] = load_frag(h1s, base_m + (lane & 15), kt, lane);
      #pragma unroll
      for (int nt = 0; nt < 8; ++nt) {
        v8f acc = {0.f,0.f,0.f,0.f,0.f,0.f,0.f,0.f};
        #pragma unroll
        for (int kt = 0; kt < 4; ++kt) {
          v16bf bfr = load_frag(Wm2T, nt*16 + (lane & 15), kt, lane);
          acc = __builtin_amdgcn_wmma_f32_16x16x32_bf16(
                  false, afr[kt], false, bfr, (short)0, acc, false, false);
        }
        int   col  = nt*16 + (lane & 15);
        float bias = bm2s[col];
        int   rb   = base_m + ((lane >> 4) << 3);
        #pragma unroll
        for (int r = 0; r < 8; ++r)
          h2s[(rb + r)*LDK + col] = f2bf(fmaxf(acc[r] + bias, 0.0f));
      }
    }
    __syncthreads();

    { // GEMM2: msgs = h2 @ Wm3 ; per-column running max (bm3 folded in later)
      v16bf afr[4];
      #pragma unroll
      for (int kt = 0; kt < 4; ++kt)
        afr[kt] = load_frag(h2s, base_m + (lane & 15), kt, lane);
      #pragma unroll
      for (int nt = 0; nt < 8; ++nt) {
        v8f acc = {0.f,0.f,0.f,0.f,0.f,0.f,0.f,0.f};
        #pragma unroll
        for (int kt = 0; kt < 4; ++kt) {
          v16bf bfr = load_frag(Wm3T, nt*16 + (lane & 15), kt, lane);
          acc = __builtin_amdgcn_wmma_f32_16x16x32_bf16(
                  false, afr[kt], false, bfr, (short)0, acc, false, false);
        }
        float pm = acc[0];
        #pragma unroll
        for (int r = 1; r < 8; ++r) pm = fmaxf(pm, acc[r]);
        mpart[(wv*2 + (lane >> 4))*128 + nt*16 + (lane & 15)] = pm;
      }
    }
    __syncthreads();

    if (tid < 128) {                            // fold 16 partials into mbuf
      float mx = mbuf[tid];
      #pragma unroll
      for (int q = 0; q < 16; ++q) mx = fmaxf(mx, mpart[q*128 + tid]);
      mbuf[tid] = mx;
    }
    __syncthreads();
  }

  if (tid < 128) m_out[bn*Hh + tid] = mbuf[tid] + bm3[tid];
}

// ---------------------------------------------------------------------------
// K3a: per-batch max over nodes of a_j = node@Wt_j, a_k = node@Wt_k  (T=8)
// ---------------------------------------------------------------------------
__global__ __launch_bounds__(256) void tripmax_kernel(
    const float* __restrict__ node, const float* __restrict__ Wt1,
    float* __restrict__ ajmax, float* __restrict__ akmax)
{
  __shared__ float red[256];
  int b = blockIdx.x, t = threadIdx.x;          // t = node index
  const float* nr = node + (b*Nn + t)*Hh;
  float aj[Tt], ak[Tt];
  #pragma unroll
  for (int tt = 0; tt < Tt; ++tt) { aj[tt] = 0.f; ak[tt] = 0.f; }
  for (int k = 0; k < Hh; ++k) {
    float nv = nr[k];
    #pragma unroll
    for (int tt = 0; tt < Tt; ++tt) {
      aj[tt] += nv * Wt1[k*Tt + tt];
      ak[tt] += nv * Wt1[(Hh + k)*Tt + tt];
    }
  }
  for (int tt = 0; tt < Tt; ++tt) {
    red[t] = aj[tt]; __syncthreads();
    for (int s = 128; s > 0; s >>= 1) {
      if (t < s) red[t] = fmaxf(red[t], red[t + s]);
      __syncthreads();
    }
    if (t == 0) ajmax[b*Tt + tt] = red[0];
    __syncthreads();
    red[t] = ak[tt]; __syncthreads();
    for (int s = 128; s > 0; s >>= 1) {
      if (t < s) red[t] = fmaxf(red[t], red[t + s]);
      __syncthreads();
    }
    if (t == 0) akmax[b*Tt + tt] = red[0];
    __syncthreads();
  }
}

// ---------------------------------------------------------------------------
// K3b: gate, readout, LayerNorm, triplet output   (per (b,n), 128 threads)
// ---------------------------------------------------------------------------
__global__ __launch_bounds__(128) void final_kernel(
    const float* __restrict__ node, const float* __restrict__ edge,
    const float* __restrict__ graph,const float* __restrict__ prev,
    const float* __restrict__ m,
    const float* __restrict__ Wg1, const float* __restrict__ bg1,
    const float* __restrict__ Wg2, const float* __restrict__ bg2,
    const float* __restrict__ Wr,  const float* __restrict__ br,
    const float* __restrict__ Wt1, const float* __restrict__ bt1,
    const float* __restrict__ Wt2, const float* __restrict__ bt2,
    const float* __restrict__ gamma,const float* __restrict__ beta,
    const float* __restrict__ ajmax,const float* __restrict__ akmax,
    float* __restrict__ out)
{
  __shared__ float sn[128], sp[128], se[128], sg[128], sm[128];
  __shared__ float r1[128], tmaxs[Tt], red[128];
  __shared__ float smu, srstd;
  int bn = blockIdx.x, t = threadIdx.x;
  int b = bn >> 8;
  sn[t] = node[bn*Hh + t];
  sp[t] = prev[bn*Hh + t];
  se[t] = edge[bn*Hh + t];
  sg[t] = graph[b*Hh + t];
  sm[t] = m[bn*Hh + t];
  __syncthreads();

  // gate stage 1: relu(gate_in @ Wg1 + bg1); rows: node,prev,m,edge,graph
  float a1 = bg1[t];
  for (int k = 0; k < Hh; ++k) {
    a1 += sn[k]*Wg1[k*Hh + t]          + sp[k]*Wg1[(Hh + k)*Hh + t]
        + sm[k]*Wg1[(2*Hh + k)*Hh + t] + se[k]*Wg1[(3*Hh + k)*Hh + t]
        + sg[k]*Wg1[(4*Hh + k)*Hh + t];
  }
  r1[t] = fmaxf(a1, 0.0f);

  // triplet pre-activations (threads 0..7)
  if (t < Tt) {
    float c = bt1[t];
    for (int k = 0; k < Hh; ++k) {
      c += sn[k]*Wt1[(2*Hh + k)*Tt + t]
         + se[k]*(Wt1[(3*Hh + k)*Tt + t] + Wt1[(4*Hh + k)*Tt + t]
                + Wt1[(5*Hh + k)*Tt + t])
         + sg[k]*Wt1[(6*Hh + k)*Tt + t];
    }
    tmaxs[t] = fmaxf(c + ajmax[b*Tt + t] + akmax[b*Tt + t], 0.0f);
  }
  __syncthreads();

  // gate stage 2 + readout
  float a2 = bg2[t];
  for (int k = 0; k < Hh; ++k) a2 += r1[k]*Wg2[k*Hh + t];
  float g = 1.0f / (1.0f + __expf(-a2));
  float hi = br[t];
  for (int k = 0; k < Hh; ++k)
    hi += sn[k]*Wr[k*Hh + t] + sp[k]*Wr[(Hh + k)*Hh + t]
        + sm[k]*Wr[(2*Hh + k)*Hh + t];
  float hg = g*hi + (1.0f - g)*sp[t];

  // LayerNorm over features
  red[t] = hg; __syncthreads();
  for (int s = 64; s > 0; s >>= 1) { if (t < s) red[t] += red[t + s]; __syncthreads(); }
  if (t == 0) smu = red[0] * (1.0f/Hh);
  __syncthreads();
  float d = hg - smu;
  red[t] = d*d; __syncthreads();
  for (int s = 64; s > 0; s >>= 1) { if (t < s) red[t] += red[t + s]; __syncthreads(); }
  if (t == 0) srstd = rsqrtf(red[0] * (1.0f/Hh) + 1e-5f);
  __syncthreads();
  out[bn*Hh + t] = d * srstd * gamma[t] + beta[t];

  // h_ij = t_max @ Wt2 + bt2
  float ho = bt2[t];
  #pragma unroll
  for (int tt = 0; tt < Tt; ++tt) ho += tmaxs[tt]*Wt2[tt*Hh + t];
  out[BNH + bn*Hh + t] = ho;
}

// ---------------------------------------------------------------------------
extern "C" void kernel_launch(void* const* d_in, const int* in_sizes, int n_in,
                              void* d_out, int out_size, void* d_ws, size_t ws_size,
                              hipStream_t stream) {
  (void)in_sizes; (void)n_in; (void)out_size; (void)ws_size;
  const float* node  = (const float*)d_in[0];
  const float* edge  = (const float*)d_in[1];
  const float* graph = (const float*)d_in[2];
  const float* prev  = (const float*)d_in[3];
  const float* Wm1 = (const float*)d_in[4];  const float* bm1 = (const float*)d_in[5];
  const float* Wm2 = (const float*)d_in[6];  const float* bm2 = (const float*)d_in[7];
  const float* Wm3 = (const float*)d_in[8];  const float* bm3 = (const float*)d_in[9];
  const float* Wg1 = (const float*)d_in[10]; const float* bg1 = (const float*)d_in[11];
  const float* Wg2 = (const float*)d_in[12]; const float* bg2 = (const float*)d_in[13];
  const float* Wr  = (const float*)d_in[14]; const float* br  = (const float*)d_in[15];
  const float* Wt1 = (const float*)d_in[16]; const float* bt1 = (const float*)d_in[17];
  const float* Wt2 = (const float*)d_in[18]; const float* bt2 = (const float*)d_in[19];
  const float* gamma = (const float*)d_in[20];
  const float* beta  = (const float*)d_in[21];

  float* wsf  = (float*)d_ws;
  float* u    = wsf;
  float* v    = wsf + BNH;
  float* m    = wsf + 2*BNH;
  float* ajmx = wsf + 3*BNH;
  float* akmx = ajmx + Bb*Tt;

  uv_kernel<<<Bb*Nn, 128, 0, stream>>>(node, prev, edge, graph, Wm1, bm1, u, v);
  tripmax_kernel<<<Bb, 256, 0, stream>>>(node, Wt1, ajmx, akmx);

  size_t smem = (size_t)(4*128*136*sizeof(__bf16))
              + (size_t)(128 + 128 + 8*2*128 + 128)*sizeof(float);
  pair_msgs_kernel<<<Bb*Nn, 256, smem, stream>>>(u, v, Wm2, bm2, Wm3, bm3, m);

  final_kernel<<<Bb*Nn, 128, 0, stream>>>(node, edge, graph, prev, m,
      Wg1, bg1, Wg2, bg2, Wr, br, Wt1, bt1, Wt2, bt2, gamma, beta,
      ajmx, akmx, (float*)d_out);
}
